// ScaledDotProductAttention_7859790152137
// MI455X (gfx1250) — matrix operand order repaired and checked by reading, not compiled, not run on hardware
//
#include <hip/hip_runtime.h>
#include <hip/hip_bf16.h>

// CDNA5 / gfx1250 flash-attention. wave32, WMMA bf16, TDM-staged K/V, DPP softmax reductions.

typedef __attribute__((ext_vector_type(16))) __bf16 v16bf;
typedef __attribute__((ext_vector_type(8)))  float  v8f;
typedef __attribute__((ext_vector_type(4)))  unsigned v4u;
typedef __attribute__((ext_vector_type(8)))  int    v8i;
typedef __attribute__((ext_vector_type(4)))  int    v4i;

#define BATCH 32
#define SEQ   2048
#define DK    128
#define QTILE 128   // query rows per block (8 waves x 16 rows)
#define KTILE 32    // keys per KV chunk
#define NWAVE 8
#define NCHUNK (SEQ / KTILE)

#if __has_builtin(__builtin_amdgcn_tensor_load_to_lds)
#define HAVE_TDM 1
#endif

union BF16Frag {
    v16bf v;
    unsigned short u[16];
    uint4 q[2];
};

__device__ __forceinline__ unsigned short f32_bf16(float f) {
    unsigned u = __builtin_bit_cast(unsigned, f);
    u += 0x7FFFu + ((u >> 16) & 1u);   // round-to-nearest-even
    return (unsigned short)(u >> 16);
}

// ---- DPP16 xor-involution reductions over 16-lane rows (no LDS traffic) ----
template <int CTRL>
__device__ __forceinline__ float dpp_mov(float x) {
    return __builtin_bit_cast(float,
        __builtin_amdgcn_update_dpp(0, __builtin_bit_cast(int, x), CTRL, 0xF, 0xF, true));
}
__device__ __forceinline__ float red_max16(float x) {
    x = fmaxf(x, dpp_mov<0xB1>(x));    // quad_perm [1,0,3,2]  (xor 1)
    x = fmaxf(x, dpp_mov<0x4E>(x));    // quad_perm [2,3,0,1]  (xor 2)
    x = fmaxf(x, dpp_mov<0x141>(x));   // row_half_mirror      (xor 7)
    x = fmaxf(x, dpp_mov<0x140>(x));   // row_mirror           (xor 15)
    return x;
}
__device__ __forceinline__ float red_sum16(float x) {
    x += dpp_mov<0xB1>(x);
    x += dpp_mov<0x4E>(x);
    x += dpp_mov<0x141>(x);
    x += dpp_mov<0x140>(x);
    return x;
}

// ---- TDM: DMA one 2D tile (KTILE rows x DK f32, contiguous) from global to LDS ----
__device__ __forceinline__ void tdm_load_2d(const void* gptr, unsigned lds_off) {
#ifdef HAVE_TDM
    unsigned long long ga = (unsigned long long)(uintptr_t)gptr;
    v4u g0;
    g0.x = 1u;                                   // count=1, user descriptor
    g0.y = lds_off;                              // lds_addr (bits 63:32)
    g0.z = (unsigned)(ga & 0xFFFFFFFFu);         // global_addr[31:0]
    g0.w = (unsigned)((ga >> 32) & 0x01FFFFFFu)  // global_addr[56:32]
         | (2u << 30);                           // D# type = 2
    v8i g1;
    g1[0] = (int)(2u << 16);                     // data_size = 4 bytes
    g1[1] = (int)((unsigned)DK << 16);           // tensor_dim0 = 128  (bits 79:48 lo)
    g1[2] = (int)((unsigned)SEQ << 16);          // dim0 hi=0 | tensor_dim1 = 2048 lo
    g1[3] = (int)((unsigned)DK << 16);           // dim1 hi=0 | tile_dim0 = 128
    g1[4] = KTILE;                               // tile_dim1 = 32 | tile_dim2 = 0
    g1[5] = DK;                                  // tensor_dim0_stride = 128 (lo 32)
    g1[6] = 0;
    g1[7] = 0;
    v4i z4 = {0, 0, 0, 0};
  #if __clang_major__ >= 23
    v8i z8 = {0, 0, 0, 0, 0, 0, 0, 0};
    __builtin_amdgcn_tensor_load_to_lds(g0, g1, z4, z4, z8, 0);
  #else
    __builtin_amdgcn_tensor_load_to_lds(g0, g1, z4, z4, 0);
  #endif
#else
    (void)gptr; (void)lds_off;
#endif
}

__device__ __forceinline__ void tdm_wait() {
#if __has_builtin(__builtin_amdgcn_s_wait_tensorcnt)
    __builtin_amdgcn_s_wait_tensorcnt(0);
#endif
}

__global__ __launch_bounds__(256)
void fa_cdna5_kernel(const float* __restrict__ q, const float* __restrict__ k,
                     const float* __restrict__ v, float* __restrict__ out) {
#ifdef HAVE_TDM
    __shared__ __align__(16) float KbF[2][KTILE * DK];   // TDM landing buffers (f32)
    __shared__ __align__(16) float VbF[2][KTILE * DK];
#endif
    __shared__ __align__(16) unsigned short Kb[KTILE * DK];      // bf16 row-major
    __shared__ __align__(16) unsigned short Vt[DK * KTILE];      // bf16 transposed [d][k]
    __shared__ __align__(16) unsigned short Pb[NWAVE][16 * KTILE];

    const int tid  = threadIdx.x;
    const int wave = tid >> 5;
    const int lane = tid & 31;
    const int half = lane >> 4;
    const int lid  = lane & 15;

    const int b    = blockIdx.y;
    const int qblk = blockIdx.x;

    const size_t bbase = (size_t)b * SEQ * DK;
    const float* kg = k + bbase;
    const float* vg = v + bbase;

    // ---- Q -> A fragments, scaled by (1/sqrt(128)) * log2(e) so softmax runs in exp2 ----
    const float qscale = 0.08838834764831845f * 1.4426950408889634f;
    const int   qrow   = qblk * QTILE + wave * 16 + lid;
    const float* qrp   = q + bbase + (size_t)qrow * DK;

    BF16Frag Af[4];
    #pragma unroll
    for (int f = 0; f < 4; ++f) {
        const int d0 = f * 32 + half * 16;
        #pragma unroll
        for (int e = 0; e < 8; ++e) {
            Af[f].u[e]     = f32_bf16(qrp[d0 + e]      * qscale);
            Af[f].u[e + 8] = f32_bf16(qrp[d0 + 8 + e] * qscale);
        }
    }

    v8f O[8];
    #pragma unroll
    for (int c = 0; c < 8; ++c) O[c] = (v8f){};
    float mrow[8], lrow[8];
    #pragma unroll
    for (int r = 0; r < 8; ++r) { mrow[r] = -3.0e38f; lrow[r] = 0.0f; }

    // ---- issue TDM DMA of one KV chunk (wave0: K, wave1: V) ----
    auto tdm_issue = [&](int buf, int chunk) {
#ifdef HAVE_TDM
        if (wave == 0)
            tdm_load_2d(kg + (size_t)chunk * KTILE * DK, (unsigned)(uintptr_t)&KbF[buf][0]);
        if (wave == 1)
            tdm_load_2d(vg + (size_t)chunk * KTILE * DK, (unsigned)(uintptr_t)&VbF[buf][0]);
#else
        (void)buf; (void)chunk;
#endif
    };

    // ---- cooperative convert: chunk f32 -> LDS bf16 (K row-major, V transposed) ----
    auto cvtstage = [&](int srcbuf, int chunk) {
#ifdef HAVE_TDM
        const float* kc = &KbF[srcbuf][0];
        const float* vc = &VbF[srcbuf][0];
#else
        (void)srcbuf;
        const float* kc = kg + (size_t)chunk * KTILE * DK;
        const float* vc = vg + (size_t)chunk * KTILE * DK;
        if (chunk + 1 < NCHUNK) {   // pull next chunk toward L2
            __builtin_prefetch(kc + KTILE * DK + tid * 16, 0, 1);
            __builtin_prefetch(vc + KTILE * DK + tid * 16, 0, 1);
        }
#endif
        #pragma unroll
        for (int i = 0; i < 4; ++i) {
            const int off  = (tid + i * 256) * 4;
            const int krow = off >> 7;
            const int d    = off & 127;
            const float4 k4 = *reinterpret_cast<const float4*>(kc + off);
            const float4 v4 = *reinterpret_cast<const float4*>(vc + off);
            uint2 kp;
            kp.x = (unsigned)f32_bf16(k4.x) | ((unsigned)f32_bf16(k4.y) << 16);
            kp.y = (unsigned)f32_bf16(k4.z) | ((unsigned)f32_bf16(k4.w) << 16);
            *reinterpret_cast<uint2*>(&Kb[off]) = kp;
            Vt[(d + 0) * KTILE + krow] = f32_bf16(v4.x);
            Vt[(d + 1) * KTILE + krow] = f32_bf16(v4.y);
            Vt[(d + 2) * KTILE + krow] = f32_bf16(v4.z);
            Vt[(d + 3) * KTILE + krow] = f32_bf16(v4.w);
        }
    };

    tdm_issue(0, 0);
    tdm_wait();
    __syncthreads();

    for (int j = 0; j < NCHUNK; ++j) {
        if (j + 1 < NCHUNK) tdm_issue((j + 1) & 1, j + 1);   // DMA next chunk during compute
        cvtstage(j & 1, j);
        __syncthreads();

        // ---- S = Q * K^T : two 16x16 f32 tiles (keys 0-15, 16-31), 4 k-steps each ----
        v8f S0 = (v8f){}, S1 = (v8f){};
        #pragma unroll
        for (int f = 0; f < 4; ++f) {
            BF16Frag B0, B1;
            const uint4* p0 = reinterpret_cast<const uint4*>(
                &Kb[(lid     ) * DK + f * 32 + half * 16]);
            const uint4* p1 = reinterpret_cast<const uint4*>(
                &Kb[(lid + 16) * DK + f * 32 + half * 16]);
            B0.q[0] = p0[0]; B0.q[1] = p0[1];
            B1.q[0] = p1[0]; B1.q[1] = p1[1];
            S0 = __builtin_amdgcn_wmma_f32_16x16x32_bf16(false, Af[f].v, false, B0.v,
                                                         (short)0, S0, false, false);
            S1 = __builtin_amdgcn_wmma_f32_16x16x32_bf16(false, Af[f].v, false, B1.v,
                                                         (short)0, S1, false, false);
        }

        // ---- online softmax (base-2), DPP reductions over 16-lane halves ----
        #pragma unroll
        for (int r = 0; r < 8; ++r) {
            const float mx    = red_max16(fmaxf(S0[r], S1[r]));
            const float mnew  = fmaxf(mrow[r], mx);
            const float alpha = exp2f(mrow[r] - mnew);
            const float p0    = exp2f(S0[r] - mnew);
            const float p1    = exp2f(S1[r] - mnew);
            mrow[r] = mnew;
            lrow[r] = lrow[r] * alpha + red_sum16(p0 + p1);

            const int row = half * 8 + r;   // C layout: VGPR r = row r / 8+r per half
            Pb[wave][row * KTILE + lid]      = f32_bf16(p0);
            Pb[wave][row * KTILE + 16 + lid] = f32_bf16(p1);

            #pragma unroll
            for (int c = 0; c < 8; ++c) O[c][r] *= alpha;
        }

        __builtin_amdgcn_wave_barrier();

        // ---- P back as A fragment (16x32 bf16) ----
        BF16Frag Pf;
        Pf.q[0] = *reinterpret_cast<const uint4*>(&Pb[wave][lid * KTILE + half * 16]);
        Pf.q[1] = *reinterpret_cast<const uint4*>(&Pb[wave][lid * KTILE + 8 + half * 16]);

        // ---- O += P * V : 8 column tiles, V from transposed LDS (contiguous 32B) ----
        #pragma unroll
        for (int c = 0; c < 8; ++c) {
            BF16Frag Bv;
            const uint4* pv = reinterpret_cast<const uint4*>(
                &Vt[(c * 16 + lid) * KTILE + half * 16]);
            Bv.q[0] = pv[0]; Bv.q[1] = pv[1];
            O[c] = __builtin_amdgcn_wmma_f32_16x16x32_bf16(false, Pf.v, false, Bv.v,
                                                           (short)0, O[c], false, false);
        }

        tdm_wait();        // TDM for chunk j+1 landed
        __syncthreads();   // safe to overwrite Kb/Vt and read next landing buffer
    }

    // ---- epilogue: normalize and store ----
    const int orow0 = qblk * QTILE + wave * 16 + half * 8;
    #pragma unroll
    for (int r = 0; r < 8; ++r) {
        const float invl = 1.0f / lrow[r];
        float* orow = out + bbase + (size_t)(orow0 + r) * DK;
        #pragma unroll
        for (int c = 0; c < 8; ++c)
            orow[c * 16 + lid] = O[c][r] * invl;
    }
}

extern "C" void kernel_launch(void* const* d_in, const int* in_sizes, int n_in,
                              void* d_out, int out_size, void* d_ws, size_t ws_size,
                              hipStream_t stream) {
    const float* q = (const float*)d_in[0];
    const float* k = (const float*)d_in[1];
    const float* v = (const float*)d_in[2];
    float* out = (float*)d_out;
    (void)in_sizes; (void)n_in; (void)out_size; (void)d_ws; (void)ws_size;

    dim3 grid(SEQ / QTILE, BATCH);   // (16, 32)
    dim3 block(256);                 // 8 wave32
    fa_cdna5_kernel<<<grid, block, 0, stream>>>(q, k, v, out);
}
